// EdgeBlock_34789235098351
// MI455X (gfx1250) — matrix operand order repaired and checked by reading, not compiled, not run on hardware
//
#include <hip/hip_runtime.h>

// ---------------------------------------------------------------------------
// EdgeBlock MLP for MI455X (gfx1250, wave32, WMMA + TDM).
// out[E,64] = relu(concat(edge_attr, x[ei1], x[ei0], u) @ W1 + b1) @ W2 + b2
// f16 WMMA 16x16x32, f32 accumulate. Weights pre-transposed/K-padded (with the
// LDS bank-padding baked in) into d_ws, then DMA'd to LDS via the CDNA5
// Tensor Data Mover (fallback: cooperative b128 copy). Each wave owns 32 edge
// rows so every weight B-fragment feeds two WMMAs. sA/sH share one LDS buffer.
// ---------------------------------------------------------------------------

typedef _Float16 v16h  __attribute__((ext_vector_type(16)));
typedef float    v8f   __attribute__((ext_vector_type(8)));
typedef unsigned u32x4 __attribute__((ext_vector_type(4)));
typedef float    f32x2 __attribute__((ext_vector_type(2)));
typedef unsigned tdm_u32x4 __attribute__((ext_vector_type(4)));
typedef int      tdm_i32x4 __attribute__((ext_vector_type(4)));
typedef int      tdm_i32x8 __attribute__((ext_vector_type(8)));

union F16Frag { v16h v; u32x4 q[2]; };

#define D_IN      208
#define D_HID     256
#define TILE_M    256            // edges per workgroup tile (8 waves * 32 rows)
#define W1P_COLS  232            // 224 K (208 padded) + 8 bank-pad halves
#define W2P_COLS  264            // 256 K + 8 bank-pad halves
#define ST_COLS   264            // shared activation buffer stride (halves)
#define W1P_ELEMS (256 * W1P_COLS)   // 59392 f16  (118784 B)
#define W2P_ELEMS (64  * W2P_COLS)   // 16896 f16  ( 33792 B)

// TDM builtin exists on both probed toolchains, with different arity:
//   ROCm 7.2 / clang-22 : 5 args (g0 u32x4, g1 i32x8, i32x4, i32x4, cpol)
//   amdgpu-toolchain / clang-23+: 6 args (..., extra i32x8, cpol)
#if __has_builtin(__builtin_amdgcn_tensor_load_to_lds) && \
    __has_builtin(__builtin_amdgcn_s_wait_tensorcnt)
#define USE_TDM 1
#endif

// ---------------------------------------------------------------------------
// Pack W1 [208,256] f32 -> w1p [256,232] f16 (N-major, K zero-padded + bank pad)
//      W2 [256, 64] f32 -> w2p [ 64,264] f16 (N-major, bank pad)
// Layouts match LDS exactly, so the LDS stage is a contiguous copy (TDM-able).
// ---------------------------------------------------------------------------
__global__ void pack_weights_kernel(const float* __restrict__ W1,
                                    const float* __restrict__ W2,
                                    _Float16* __restrict__ w1p,
                                    _Float16* __restrict__ w2p) {
    int i = blockIdx.x * blockDim.x + threadIdx.x;
    if (i < W1P_ELEMS) {
        int n = i / W1P_COLS, k = i % W1P_COLS;
        w1p[i] = (k < D_IN) ? (_Float16)W1[k * 256 + n] : (_Float16)0.f;
    }
    if (i < W2P_ELEMS) {
        int n = i / W2P_COLS, k = i % W2P_COLS;
        w2p[i] = (k < 256) ? (_Float16)W2[k * 64 + n] : (_Float16)0.f;
    }
}

__device__ __forceinline__ void st2h(_Float16* p, f32x2 v) {
    union { _Float16 h[2]; unsigned w; } t;
    t.h[0] = (_Float16)v.x;
    t.h[1] = (_Float16)v.y;
    *(unsigned*)p = t.w;
}

#ifdef USE_TDM
// Contiguous global->LDS DMA of `ndw` dwords (<= 65535) via the TDM.
// D# built per CDNA5 ISA: group0 {count=1, lds_addr,
// global_addr, type=2}; group1 {data_size=4B, 1-row tile of ndw dwords}.
__device__ __forceinline__ void tdm_copy_to_lds(void* lds_dst, const void* gsrc,
                                                unsigned ndw) {
    unsigned lds_addr = (unsigned)(unsigned long long)lds_dst; // low 32b = LDS offset
    unsigned long long ga = (unsigned long long)gsrc;
    tdm_u32x4 g0;
    g0.x = 0x1u;                                          // count=1, user D#
    g0.y = lds_addr;                                      // [63:32]  lds_addr
    g0.z = (unsigned)(ga & 0xFFFFFFFFu);                  // [95:64]  gaddr lo
    g0.w = (unsigned)((ga >> 32) & 0x01FFFFFFu) | (2u << 30); // gaddr hi | type=2
    tdm_i32x8 g1;
    g1[0] = (int)(2u << 16);                  // wg_mask=0, data_size=2 (4 bytes)
    g1[1] = (int)((ndw & 0xFFFFu) << 16);     // tensor_dim0[15:0]
    g1[2] = (int)((ndw >> 16) | (1u << 16));  // tensor_dim0[31:16], tensor_dim1=1
    g1[3] = (int)((ndw & 0xFFFFu) << 16);     // tensor_dim1 hi=0, tile_dim0=ndw
    g1[4] = 1;                                // tile_dim1=1, tile_dim2=0
    g1[5] = (int)ndw;                         // tensor_dim0_stride[31:0]
    g1[6] = 0;                                // stride hi / dim1_stride lo
    g1[7] = 0;
    tdm_i32x4 z4 = {0, 0, 0, 0};
#if defined(__clang_major__) && (__clang_major__ >= 23)
    tdm_i32x8 z8 = {0, 0, 0, 0, 0, 0, 0, 0};
    __builtin_amdgcn_tensor_load_to_lds(g0, g1, z4, z4, z8, 0);
#else
    __builtin_amdgcn_tensor_load_to_lds(g0, g1, z4, z4, 0);
#endif
}
#endif

// ---------------------------------------------------------------------------
// Main fused kernel: 256 threads = 8 waves; each wave owns 32 edge rows.
// ---------------------------------------------------------------------------
__global__ __launch_bounds__(256, 1)
void edge_mlp_wmma(const float* __restrict__ x,
                   const float* __restrict__ ea,
                   const float* __restrict__ u,
                   const float* __restrict__ b1,
                   const float* __restrict__ b2,
                   const long long* __restrict__ ei,    // [2, E] int64
                   const _Float16* __restrict__ w1p,
                   const _Float16* __restrict__ w2p,
                   float* __restrict__ out,
                   int n_edges, int tiles) {
    // ---- LDS (CDNA5: 320KB/WG). Total ~290KB. ----
    __shared__ _Float16 sW1[256][W1P_COLS];      // 118784 B  W1^T f16
    __shared__ _Float16 sW2[64][W2P_COLS];       //  33792 B  W2^T f16
    __shared__ _Float16 sT[8][32][ST_COLS];      // 135168 B  features / hidden (aliased)
    __shared__ float    sB1[256];
    __shared__ float    sB2[64];

    const int tid = threadIdx.x;

    // ---- stage weights into LDS ----
#ifdef USE_TDM
    if (tid == 0) {
        tdm_copy_to_lds(&sW1[0][0], w1p, (W1P_ELEMS * 2) / 4);   // 29696 dwords
        tdm_copy_to_lds(&sW2[0][0], w2p, (W2P_ELEMS * 2) / 4);   //  8448 dwords
        __builtin_amdgcn_s_wait_tensorcnt(0);
    }
#else
    for (int i = tid; i < W1P_ELEMS / 8; i += 256)
        *(u32x4*)((_Float16*)sW1 + i * 8) = *(const u32x4*)(w1p + i * 8);
    for (int i = tid; i < W2P_ELEMS / 8; i += 256)
        *(u32x4*)((_Float16*)sW2 + i * 8) = *(const u32x4*)(w2p + i * 8);
#endif
    sB1[tid] = b1[tid];
    if (tid < 64) sB2[tid] = b2[tid];
    __syncthreads();

    const int wave = tid >> 5;
    const int lane = tid & 31;
    const int m    = lane & 15;   // A row or B/D col index within a 16-tile
    const int h    = lane >> 4;   // half-wave selector

    for (int t = blockIdx.x; t < tiles; t += gridDim.x) {
        const int row0 = t * TILE_M + wave * 32;

        // ---- stage collected features [ea | recv | send | u | 0pad] as f16 ----
        for (int r = 0; r < 32; ++r) {
            int g = row0 + r;
            if (g >= n_edges) g = n_edges - 1;        // tail clamp
            long long snd = ei[g];                    // edge_index[0] (sender)
            long long rcv = ei[(size_t)n_edges + g];  // edge_index[1] (receiver)
            f32x2 e2 = *(const f32x2*)(ea + (size_t)g   * 64 + 2 * lane);
            f32x2 r2 = *(const f32x2*)(x  + (size_t)rcv * 64 + 2 * lane);
            f32x2 s2 = *(const f32x2*)(x  + (size_t)snd * 64 + 2 * lane);
            st2h(&sT[wave][r][      2 * lane], e2);
            st2h(&sT[wave][r][ 64 + 2 * lane], r2);
            st2h(&sT[wave][r][128 + 2 * lane], s2);
            if (lane < 8)
                st2h(&sT[wave][r][192 + 2 * lane], *(const f32x2*)(u + 2 * lane));
            else if (lane < 16)                       // re-zero K-pad cols 208..223
                *(unsigned*)&sT[wave][r][D_IN + 2 * (lane - 8)] = 0u;
            // prefetch next grid-stride tile's edge_attr row
            int gn = g + gridDim.x * TILE_M;
            if (gn < n_edges)
                __builtin_prefetch(ea + (size_t)gn * 64 + 2 * lane, 0, 1);
        }

        // ---- lift A fragments into registers (2 M-tiles x 7 K-chunks) ----
        // A elem i -> K = 32*kc + i%8 + 8*h + 16*(i/8)
        F16Frag a[2][7];
#pragma unroll
        for (int mt = 0; mt < 2; ++mt)
#pragma unroll
            for (int kc = 0; kc < 7; ++kc) {
                a[mt][kc].q[0] = *(const u32x4*)&sT[wave][mt * 16 + m][32 * kc +      8 * h];
                a[mt][kc].q[1] = *(const u32x4*)&sT[wave][mt * 16 + m][32 * kc + 16 + 8 * h];
            }

        // ---- GEMM1: hidden[32,256] = relu(A @ W1 + b1) -> sT (f16, in place) ----
        for (int nt = 0; nt < 16; ++nt) {
            v8f acc0 = {0.f, 0.f, 0.f, 0.f, 0.f, 0.f, 0.f, 0.f};
            v8f acc1 = {0.f, 0.f, 0.f, 0.f, 0.f, 0.f, 0.f, 0.f};
            const _Float16* wrow = &sW1[nt * 16 + m][0];   // B: N=m, K = 8h + s, 16 + 8h + (s-8)
#pragma unroll
            for (int kc = 0; kc < 7; ++kc) {
                F16Frag b;
                b.q[0] = *(const u32x4*)(wrow + 32 * kc +      8 * h);
                b.q[1] = *(const u32x4*)(wrow + 32 * kc + 16 + 8 * h);
                acc0 = __builtin_amdgcn_wmma_f32_16x16x32_f16(
                    false, a[0][kc].v, false, b.v, (short)0, acc0, false, false);
                acc1 = __builtin_amdgcn_wmma_f32_16x16x32_f16(
                    false, a[1][kc].v, false, b.v, (short)0, acc1, false, false);
            }
            const float bias = sB1[nt * 16 + m];
#pragma unroll
            for (int r = 0; r < 8; ++r) {   // D elem r -> row r+8h, col nt*16+m
                float v0 = acc0[r] + bias;
                float v1 = acc1[r] + bias;
                v0 = v0 > 0.f ? v0 : 0.f;
                v1 = v1 > 0.f ? v1 : 0.f;
                sT[wave][     r + 8 * h][nt * 16 + m] = (_Float16)v0;
                sT[wave][16 + r + 8 * h][nt * 16 + m] = (_Float16)v1;
            }
        }

        // ---- GEMM2: out[32,64] = hidden @ W2 + b2 ----
        v8f acc2[2][4];
#pragma unroll
        for (int mt = 0; mt < 2; ++mt)
#pragma unroll
            for (int nt = 0; nt < 4; ++nt)
                acc2[mt][nt] = (v8f){0.f, 0.f, 0.f, 0.f, 0.f, 0.f, 0.f, 0.f};
#pragma unroll
        for (int kc = 0; kc < 8; ++kc) {
            F16Frag ah0, ah1;
            ah0.q[0] = *(const u32x4*)&sT[wave][     m][32 * kc +      8 * h];
            ah0.q[1] = *(const u32x4*)&sT[wave][     m][32 * kc + 16 + 8 * h];
            ah1.q[0] = *(const u32x4*)&sT[wave][16 + m][32 * kc +      8 * h];
            ah1.q[1] = *(const u32x4*)&sT[wave][16 + m][32 * kc + 16 + 8 * h];
#pragma unroll
            for (int nt = 0; nt < 4; ++nt) {
                const _Float16* wrow = &sW2[nt * 16 + m][0];
                F16Frag b;
                b.q[0] = *(const u32x4*)(wrow + 32 * kc +      8 * h);
                b.q[1] = *(const u32x4*)(wrow + 32 * kc + 16 + 8 * h);
                acc2[0][nt] = __builtin_amdgcn_wmma_f32_16x16x32_f16(
                    false, ah0.v, false, b.v, (short)0, acc2[0][nt], false, false);
                acc2[1][nt] = __builtin_amdgcn_wmma_f32_16x16x32_f16(
                    false, ah1.v, false, b.v, (short)0, acc2[1][nt], false, false);
            }
        }

        // ---- store output (f32, coalesced across m) ----
#pragma unroll
        for (int mt = 0; mt < 2; ++mt)
#pragma unroll
            for (int nt = 0; nt < 4; ++nt) {
                const float bias = sB2[nt * 16 + m];
#pragma unroll
                for (int r = 0; r < 8; ++r) {
                    int grow = row0 + mt * 16 + r + 8 * h;
                    if (grow < n_edges)
                        out[(size_t)grow * 64 + nt * 16 + m] = acc2[mt][nt][r] + bias;
                }
            }
    }
}

// ---------------------------------------------------------------------------
// Host launcher
// ---------------------------------------------------------------------------
extern "C" void kernel_launch(void* const* d_in, const int* in_sizes, int n_in,
                              void* d_out, int out_size, void* d_ws, size_t ws_size,
                              hipStream_t stream) {
    const float*     x_ = (const float*)d_in[0];      // [50000, 64]
    const float*     ea = (const float*)d_in[1];      // [E, 64]
    const float*     u_ = (const float*)d_in[2];      // [1, 16]
    const float*     W1 = (const float*)d_in[3];      // [208, 256]
    const float*     b1 = (const float*)d_in[4];      // [256]
    const float*     W2 = (const float*)d_in[5];      // [256, 64]
    const float*     b2 = (const float*)d_in[6];      // [64]
    const long long* ei = (const long long*)d_in[7];  // [2, E] int64

    const int n_edges = in_sizes[1] / 64;

    _Float16* w1p = (_Float16*)d_ws;
    _Float16* w2p = (_Float16*)((char*)d_ws + (size_t)W1P_ELEMS * sizeof(_Float16));
    (void)ws_size; (void)n_in; (void)out_size;

    {
        int grid = (W1P_ELEMS + 255) / 256;   // covers W2P_ELEMS too
        pack_weights_kernel<<<grid, 256, 0, stream>>>(W1, W2, w1p, w2p);
    }

    const int tiles = (n_edges + TILE_M - 1) / TILE_M;   // 3125 for E=800000
    int grid = tiles < 1024 ? tiles : 1024;
    edge_mlp_wmma<<<grid, 256, 0, stream>>>(x_, ea, u_, b1, b2, ei,
                                            w1p, w2p, (float*)d_out,
                                            n_edges, tiles);
}
